// LongformerBlock_58652073394313
// MI455X (gfx1250) — matrix operand order repaired and checked by reading, not compiled, not run on hardware
//
#include <hip/hip_runtime.h>
#include <hip/hip_bf16.h>

// ---------------- CDNA5 WMMA types ----------------
typedef __attribute__((ext_vector_type(16))) __bf16 v16bf;
typedef __attribute__((ext_vector_type(8)))  float  v8f;

static __device__ __forceinline__ unsigned short f2b(float f) {
    union { float f; unsigned u; } v; v.f = f;
    unsigned r = v.u + 0x7FFFu + ((v.u >> 16) & 1u);   // round-to-nearest-even
    return (unsigned short)(r >> 16);
}

static __device__ __forceinline__ v8f wmma_bf16(v16bf a, v16bf b, v8f c) {
    // D = A(16x32 bf16) * B(32x16 bf16) + C(16x16 f32)
    return __builtin_amdgcn_wmma_f32_16x16x32_bf16(false, a, false, b, (short)0, c, false, false);
}

union Frag16 { v16bf bf; unsigned u[8]; };

// ---------------- f32 -> bf16 conversion ----------------
__global__ void cvt_f32_bf16(const float* __restrict__ in, unsigned short* __restrict__ out, int n) {
    int i = blockIdx.x * blockDim.x + threadIdx.x;
    if (i < n) out[i] = f2b(in[i]);
}

// ---------------- Tiled WMMA GEMM ----------------
// C[M,N] = A[M,K](bf16,row) * B[K,N](bf16,row) + bias; optional ReLU; f32 and/or bf16 out.
// Block: 256 threads (8 waves). Tile: 64x64, K-step 64.
// A tile staged with GLOBAL_LOAD_ASYNC_TO_LDS_B128 (ASYNCcnt); B transposed via VGPRs.
__global__ __launch_bounds__(256) void gemm_bf16_kernel(
    const unsigned short* __restrict__ A, const unsigned short* __restrict__ Bm,
    const float* __restrict__ bias, float* __restrict__ outf, unsigned short* __restrict__ outb,
    int M, int N, int K, int relu)
{
    __shared__ unsigned short Alds[64 * 64];   // [row][k]
    __shared__ unsigned short Blds[64 * 64];   // [n][k]  (transposed on fill)
    const int tid  = threadIdx.x;
    const int wave = tid >> 5, lane = tid & 31;
    const int half = lane >> 4, lc = lane & 15;
    const int wm = wave & 3, wg = wave >> 2;
    const int nblk = N >> 6;
    const int bm = blockIdx.x / nblk, bn = blockIdx.x % nblk;

    v8f acc0 = {0.f,0.f,0.f,0.f,0.f,0.f,0.f,0.f};
    v8f acc1 = acc0;

    const int arow = tid >> 2;            // 0..63
    const int acol = (tid & 3) * 16;      // 16 bf16 (32B) per thread
    const unsigned aoff0 = (unsigned)(uintptr_t)&Alds[arow * 64 + acol];
    const unsigned aoff1 = aoff0 + 16;

    const int bkr = tid >> 2;             // 0..63 (k row of B)
    const int bnc = (tid & 3) * 16;       // 16 n values per thread

    for (int kk = 0; kk < K; kk += 64) {
        // ---- A tile: async copy global -> LDS (row-major, 2x16B per thread) ----
        const unsigned short* asrc = A + (size_t)(bm * 64 + arow) * K + kk + acol;
        unsigned long long ga0 = (unsigned long long)(uintptr_t)asrc;
        unsigned long long ga1 = ga0 + 16ull;
        asm volatile("global_load_async_to_lds_b128 %0, %1, off"
                     :: "v"(aoff0), "v"(ga0) : "memory");
        asm volatile("global_load_async_to_lds_b128 %0, %1, off"
                     :: "v"(aoff1), "v"(ga1) : "memory");

        // ---- B tile: 16 contiguous bf16 per thread, transpose-scatter into Blds[n][k] ----
        const unsigned short* bsrc = Bm + (size_t)(kk + bkr) * N + bn * 64 + bnc;
        uint4 b0 = *reinterpret_cast<const uint4*>(bsrc);
        uint4 b1 = *reinterpret_cast<const uint4*>(bsrc + 8);
        {
            unsigned short e[16] = {
                (unsigned short)(b0.x), (unsigned short)(b0.x >> 16),
                (unsigned short)(b0.y), (unsigned short)(b0.y >> 16),
                (unsigned short)(b0.z), (unsigned short)(b0.z >> 16),
                (unsigned short)(b0.w), (unsigned short)(b0.w >> 16),
                (unsigned short)(b1.x), (unsigned short)(b1.x >> 16),
                (unsigned short)(b1.y), (unsigned short)(b1.y >> 16),
                (unsigned short)(b1.z), (unsigned short)(b1.z >> 16),
                (unsigned short)(b1.w), (unsigned short)(b1.w >> 16) };
            #pragma unroll
            for (int i = 0; i < 16; ++i) Blds[(bnc + i) * 64 + bkr] = e[i];
        }

        if (kk + 64 < K) {                 // prefetch next K-step
            __builtin_prefetch(asrc + 64, 0, 1);
            __builtin_prefetch(bsrc + (size_t)64 * N, 0, 1);
        }

        asm volatile("s_wait_asynccnt 0" ::: "memory");
        __syncthreads();

        const unsigned* ap = reinterpret_cast<const unsigned*>(Alds);
        const unsigned* bp = reinterpret_cast<const unsigned*>(Blds);
        const int n0 = (wg * 2 + 0) * 16 + lc;
        const int n1 = (wg * 2 + 1) * 16 + lc;
        #pragma unroll
        for (int ks = 0; ks < 2; ++ks) {
            Frag16 fa, fb0, fb1;
            #pragma unroll
            for (int v = 0; v < 8; ++v) {
                int kb2 = v + 8 * half;                      // A layout: K 16-blocks by lane-half
                fa.u[v] = ap[(wm * 16 + lc) * 32 + ks * 16 + kb2];
            }
            #pragma unroll
            for (int v = 0; v < 8; ++v) {                    // B layout: K 16-blocks by lane-half
                fb0.u[v] = bp[n0 * 32 + ks * 16 + v + 8 * half];
                fb1.u[v] = bp[n1 * 32 + ks * 16 + v + 8 * half];
            }
            acc0 = wmma_bf16(fa.bf, fb0.bf, acc0);
            acc1 = wmma_bf16(fa.bf, fb1.bf, acc1);
        }
        __syncthreads();
    }

    #pragma unroll
    for (int i = 0; i < 2; ++i) {
        v8f a = i ? acc1 : acc0;
        const int nt = wg * 2 + i;
        #pragma unroll
        for (int r = 0; r < 8; ++r) {
            int m = bm * 64 + wm * 16 + r + 8 * half;
            int n = bn * 64 + nt * 16 + lc;
            float val = a[r] + (bias ? bias[n] : 0.f);
            if (relu) val = fmaxf(val, 0.f);
            size_t idx = (size_t)m * N + n;
            if (outf) outf[idx] = val;
            if (outb) outb[idx] = f2b(val);
        }
    }
}

// ---------------- Banded flash-attention ----------------
// One wave per (batch, head, 16-query tile). Keys covered in 9 pair-tiles of 32
// spanning [q0-128, q0+159] (band is +/-128). Online softmax, f32 O accumulators.
__global__ __launch_bounds__(32) void attn_kernel(
    const unsigned short* __restrict__ Q, const unsigned short* __restrict__ Kx,
    const unsigned short* __restrict__ V, unsigned short* __restrict__ O)
{
    constexpr int L = 4096, Dm = 512, HD = 64, WIN = 128;
    __shared__ unsigned short Plds[16 * 32];
    const int lane = threadIdx.x, half = lane >> 4, lc = lane & 15;
    const int qt = blockIdx.x & 255;            // L/16
    const int hh = (blockIdx.x >> 8) & 7;       // head
    const int b  = blockIdx.x >> 11;            // batch
    const int q0 = qt * 16;
    const int rb = b * L;
    const float scale = 0.125f;                 // 1/sqrt(64)

    // Q A-fragments (HD=64 -> two K=32 fragments), rows = lane&15
    Frag16 qf[2];
    const unsigned* qp = reinterpret_cast<const unsigned*>(Q);
    #pragma unroll
    for (int kk = 0; kk < 2; ++kk) {
        int base2 = ((rb + q0 + lc) * Dm + hh * HD + kk * 32) >> 1;
        #pragma unroll
        for (int v = 0; v < 8; ++v) qf[kk].u[v] = qp[base2 + v + 8 * half];
    }

    v8f z = {0.f,0.f,0.f,0.f,0.f,0.f,0.f,0.f};
    v8f o[4]; o[0] = z; o[1] = z; o[2] = z; o[3] = z;
    float rowmax[8], rowsum[8];
    #pragma unroll
    for (int r = 0; r < 8; ++r) { rowmax[r] = -1e30f; rowsum[r] = 0.f; }

    const unsigned* kp = reinterpret_cast<const unsigned*>(Kx);

    for (int t = 0; t < 9; ++t) {
        const int kt0 = q0 - WIN + 32 * t;
        v8f s0 = z, s1 = z;
        #pragma unroll
        for (int sub = 0; sub < 2; ++sub) {
            int key = kt0 + sub * 16 + lc;
            int ck  = min(max(key, 0), L - 1);
            #pragma unroll
            for (int kk = 0; kk < 2; ++kk) {
                Frag16 kf;
                int base2 = ((rb + ck) * Dm + hh * HD + kk * 32) >> 1;
                #pragma unroll
                for (int v = 0; v < 8; ++v) kf.u[v] = kp[base2 + v + 8 * half];
                if (sub == 0) s0 = wmma_bf16(qf[kk].bf, kf.bf, s0);
                else          s1 = wmma_bf16(qf[kk].bf, kf.bf, s1);
            }
        }
        // mask + scale
        float sm0[8], sm1[8];
        #pragma unroll
        for (int r = 0; r < 8; ++r) {
            int q   = q0 + r + 8 * half;
            int k0i = kt0 + lc;
            int k1i = kt0 + 16 + lc;
            int d0  = q - k0i, d1 = q - k1i;
            bool ok0 = (k0i >= 0) && (k0i < L) && (d0 <= WIN) && (d0 >= -WIN);
            bool ok1 = (k1i >= 0) && (k1i < L) && (d1 <= WIN) && (d1 >= -WIN);
            sm0[r] = ok0 ? s0[r] * scale : -1e30f;
            sm1[r] = ok1 ? s1[r] * scale : -1e30f;
        }
        // online softmax: row reductions over the 16 lanes of this half-wave
        #pragma unroll
        for (int r = 0; r < 8; ++r) {
            float tm = fmaxf(sm0[r], sm1[r]);
            #pragma unroll
            for (int m = 1; m < 16; m <<= 1) tm = fmaxf(tm, __shfl_xor(tm, m, 16));
            float nm  = fmaxf(rowmax[r], tm);
            float esc = __expf(rowmax[r] - nm);
            float p0  = (sm0[r] > -5e29f) ? __expf(sm0[r] - nm) : 0.f;
            float p1  = (sm1[r] > -5e29f) ? __expf(sm1[r] - nm) : 0.f;
            float ts  = p0 + p1;
            #pragma unroll
            for (int m = 1; m < 16; m <<= 1) ts += __shfl_xor(ts, m, 16);
            rowsum[r] = rowsum[r] * esc + ts;
            rowmax[r] = nm;
            #pragma unroll
            for (int j = 0; j < 4; ++j) o[j][r] *= esc;
            int mrow = r + 8 * half;
            Plds[mrow * 32 + lc]      = f2b(p0);
            Plds[mrow * 32 + 16 + lc] = f2b(p1);
        }
        __syncthreads();
        // re-lay P out as a 16x32 A-fragment via LDS
        Frag16 pf;
        const unsigned* pp = reinterpret_cast<const unsigned*>(Plds);
        #pragma unroll
        for (int v = 0; v < 8; ++v) pf.u[v] = pp[lc * 16 + v + 8 * half];
        // O += P * V  (4 column blocks of HD)
        #pragma unroll
        for (int j = 0; j < 4; ++j) {
            Frag16 vf;
            #pragma unroll
            for (int v = 0; v < 8; ++v) {
                int kA = kt0 + 2 * v + 16 * half;
                int cA = min(max(kA, 0), L - 1);
                int cB = min(max(kA + 1, 0), L - 1);
                unsigned e0 = V[(size_t)(rb + cA) * Dm + hh * HD + j * 16 + lc];
                unsigned e1 = V[(size_t)(rb + cB) * Dm + hh * HD + j * 16 + lc];
                vf.u[v] = e0 | (e1 << 16);
            }
            o[j] = wmma_bf16(pf.bf, vf.bf, o[j]);
        }
        __syncthreads();
    }

    #pragma unroll
    for (int r = 0; r < 8; ++r) {
        float inv = rowsum[r] > 0.f ? 1.f / rowsum[r] : 0.f;
        #pragma unroll
        for (int j = 0; j < 4; ++j) {
            float val = o[j][r] * inv;
            O[(size_t)(rb + q0 + r + 8 * half) * Dm + hh * HD + j * 16 + lc] = f2b(val);
        }
    }
}

// ---------------- residual + LayerNorm (D=512) ----------------
__global__ __launch_bounds__(256) void add_ln_kernel(
    const float* __restrict__ a, const float* __restrict__ b,
    const float* __restrict__ g, const float* __restrict__ be,
    float* __restrict__ outf, unsigned short* __restrict__ outb)
{
    __shared__ float red[256];
    const int row = blockIdx.x, tid = threadIdx.x;
    const size_t base = (size_t)row * 512;
    float v0 = a[base + tid] + b[base + tid];
    float v1 = a[base + tid + 256] + b[base + tid + 256];
    red[tid] = v0 + v1; __syncthreads();
    for (int s = 128; s > 0; s >>= 1) { if (tid < s) red[tid] += red[tid + s]; __syncthreads(); }
    float mean = red[0] * (1.f / 512.f);
    __syncthreads();
    float d0 = v0 - mean, d1 = v1 - mean;
    red[tid] = d0 * d0 + d1 * d1; __syncthreads();
    for (int s = 128; s > 0; s >>= 1) { if (tid < s) red[tid] += red[tid + s]; __syncthreads(); }
    float rstd = rsqrtf(red[0] * (1.f / 512.f) + 1e-5f);
    float o0 = d0 * rstd * g[tid] + be[tid];
    float o1 = d1 * rstd * g[tid + 256] + be[tid + 256];
    if (outf) { outf[base + tid] = o0; outf[base + tid + 256] = o1; }
    if (outb) { outb[base + tid] = f2b(o0); outb[base + tid + 256] = f2b(o1); }
}

// ---------------- host side ----------------
extern "C" void kernel_launch(void* const* d_in, const int* in_sizes, int n_in,
                              void* d_out, int out_size, void* d_ws, size_t ws_size,
                              hipStream_t stream)
{
    (void)in_sizes; (void)n_in; (void)out_size; (void)ws_size;
    const float* x   = (const float*)d_in[0];
    const float* Wq  = (const float*)d_in[1];
    const float* bq  = (const float*)d_in[2];
    const float* Wk  = (const float*)d_in[3];
    const float* bk  = (const float*)d_in[4];
    const float* Wv  = (const float*)d_in[5];
    const float* bv  = (const float*)d_in[6];
    const float* Wo  = (const float*)d_in[7];
    const float* bo  = (const float*)d_in[8];
    const float* g1  = (const float*)d_in[9];
    const float* be1 = (const float*)d_in[10];
    const float* W1  = (const float*)d_in[11];
    const float* b1  = (const float*)d_in[12];
    const float* W2  = (const float*)d_in[13];
    const float* b2  = (const float*)d_in[14];
    const float* g2  = (const float*)d_in[15];
    const float* be2 = (const float*)d_in[16];

    constexpr int NT = 8192, D = 512, FF = 2048;

    char* ws = (char*)d_ws; size_t off = 0;
    auto alloc = [&](size_t bytes) -> void* {
        void* p = ws + off; off = (off + bytes + 255) & ~(size_t)255; return p;
    };
    unsigned short* xb   = (unsigned short*)alloc((size_t)NT * D * 2);
    unsigned short* wqb  = (unsigned short*)alloc((size_t)D * D * 2);
    unsigned short* wkb  = (unsigned short*)alloc((size_t)D * D * 2);
    unsigned short* wvb  = (unsigned short*)alloc((size_t)D * D * 2);
    unsigned short* wob  = (unsigned short*)alloc((size_t)D * D * 2);
    unsigned short* w1b  = (unsigned short*)alloc((size_t)D * FF * 2);
    unsigned short* w2b  = (unsigned short*)alloc((size_t)FF * D * 2);
    unsigned short* qb   = (unsigned short*)alloc((size_t)NT * D * 2);
    unsigned short* kb   = (unsigned short*)alloc((size_t)NT * D * 2);
    unsigned short* vb   = (unsigned short*)alloc((size_t)NT * D * 2);
    unsigned short* ab   = (unsigned short*)alloc((size_t)NT * D * 2);
    float*          attp = (float*)alloc((size_t)NT * D * 4);
    float*          hf   = (float*)alloc((size_t)NT * D * 4);
    unsigned short* hb   = (unsigned short*)alloc((size_t)NT * D * 2);
    unsigned short* ffb  = (unsigned short*)alloc((size_t)NT * FF * 2);
    float*          ff2  = (float*)alloc((size_t)NT * D * 4);

    auto cvt = [&](const float* s, unsigned short* d, int n) {
        cvt_f32_bf16<<<(n + 255) / 256, 256, 0, stream>>>(s, d, n);
    };
    cvt(x, xb, NT * D);
    cvt(Wq, wqb, D * D); cvt(Wk, wkb, D * D); cvt(Wv, wvb, D * D); cvt(Wo, wob, D * D);
    cvt(W1, w1b, D * FF); cvt(W2, w2b, FF * D);

    auto gemm = [&](const unsigned short* A, const unsigned short* Bm, const float* bias,
                    float* outf, unsigned short* outb, int M, int N, int K, int relu) {
        gemm_bf16_kernel<<<dim3((M / 64) * (N / 64)), 256, 0, stream>>>(A, Bm, bias, outf, outb, M, N, K, relu);
    };

    gemm(xb, wqb, bq, nullptr, qb, NT, D, D, 0);
    gemm(xb, wkb, bk, nullptr, kb, NT, D, D, 0);
    gemm(xb, wvb, bv, nullptr, vb, NT, D, D, 0);

    attn_kernel<<<4096, 32, 0, stream>>>(qb, kb, vb, ab);

    gemm(ab, wob, bo, attp, nullptr, NT, D, D, 0);
    add_ln_kernel<<<NT, 256, 0, stream>>>(x, attp, g1, be1, hf, hb);

    gemm(hb, w1b, b1, nullptr, ffb, NT, FF, D, 1);
    gemm(ffb, w2b, b2, ff2, nullptr, NT, D, FF, 0);
    add_ln_kernel<<<NT, 256, 0, stream>>>(hf, ff2, g2, be2, (float*)d_out, nullptr);
}
